// Gemma2Attention_58978490908721
// MI455X (gfx1250) — compile-verified, matrix operand order checked by reading, not run on hardware
//
#include <hip/hip_runtime.h>
#include <hip/hip_bf16.h>

#define H_Q   8
#define KVH_  4
#define DH    256
#define HID_  2304
#define SEQ   4096
#define SWIN  2048
#define QDIM  (H_Q * DH)    // 2048
#define KVDIM (KVH_ * DH)   // 1024
#define SCALE_   0.0625f    // 256^-0.5
#define SOFTCAP_ 50.0f

#define KLD 264   // LDS row stride for K tile (bank-conflict-free, 16B aligned)
#define VLD 40    // LDS row stride for V^T tile
#define PLD 40    // LDS row stride for P tile

typedef __attribute__((ext_vector_type(16))) __bf16 v16bf;
typedef __attribute__((ext_vector_type(8)))  __bf16 v8bf;
typedef __attribute__((ext_vector_type(8)))  float  v8f;

__device__ __forceinline__ v8f wmma_bf16(v16bf a, v16bf b, v8f c) {
  // D = A(16x32 bf16) * B(32x16 bf16) + C(16x16 f32)
  return __builtin_amdgcn_wmma_f32_16x16x32_bf16(
      false, a, false, b, (short)0, c, false, false);
}

__device__ __forceinline__ float fast_tanh(float x) {
  // tanh(x) = (e^2x - 1) / (e^2x + 1); hardware exp + rcp
  x = fminf(12.0f, fmaxf(-12.0f, x));
  float e = __expf(2.0f * x);
  return (e - 1.0f) * __builtin_amdgcn_rcpf(e + 1.0f);
}

// ---- TDM: DMA one 32x256 bf16 tile (row stride KVDIM elems) into LDS ----
// D# per CDNA5 ISA 8.3/8.4; pad_interval=128 DWORDs (one 512B row),
// pad_amount=4 DWORDs -> LDS row stride 264 bf16 == KLD.
// This toolchain exposes the 6-arg builtin form:
//   (uint32x4 g0, int32x8 g1, int32x4 g2, int32x4 g3, int32x8, i32 cpol)
#if __has_builtin(__builtin_amdgcn_tensor_load_to_lds) && \
    __has_builtin(__builtin_amdgcn_s_wait_tensorcnt)
#define HAVE_TDM 1
typedef __attribute__((ext_vector_type(4))) unsigned int v4u;
typedef __attribute__((ext_vector_type(8))) int v8i_;
typedef __attribute__((ext_vector_type(4))) int v4i_;
__device__ __forceinline__ void tdm_load_tile(const __bf16* gptr, unsigned lds_off) {
  unsigned long long ga = (unsigned long long)(uintptr_t)gptr;
  v4u g0;
  g0[0] = 1u;                                   // count=1 (valid user D#)
  g0[1] = lds_off;                              // LDS byte address of tile
  g0[2] = (unsigned)ga;                         // global_addr[31:0]
  g0[3] = (unsigned)(ga >> 32) | (2u << 30);    // global_addr[56:32] | type=2
  v8i_ g1;
  g1[0] = 0x10000                               // data_size=1 (2 bytes)
        | (1 << 20)                             // pad_enable
        | (6 << 22)                             // pad_interval: 2<<6=128 DWORDs
        | (3 << 25);                            // pad_amount: 4 DWORDs
  g1[1] = 0;                                    // tensor_dim0 = 1<<30 (lo16)
  g1[2] = 0x4000;                               // tensor_dim0 hi16 | dim1 lo16
  g1[3] = 0x4000 | (256 << 16);                 // dim1 hi16 | tile_dim0=256
  g1[4] = 32;                                   // tile_dim1=32, tile_dim2=0
  g1[5] = KVDIM;                                // tensor_dim0_stride = 1024
  g1[6] = 0;
  g1[7] = 0;
  v4i_ z4 = {0, 0, 0, 0};                       // groups 2/3 unused (2D tile)
  v8i_ z8 = {0, 0, 0, 0, 0, 0, 0, 0};
  __builtin_amdgcn_tensor_load_to_lds(g0, g1, z4, z4, z8, 0);
}
#else
#define HAVE_TDM 0
#endif

// ---- fragment loaders (layouts per CDNA5 ISA 7.12.2) --------------------
// lane l<16 -> row/col index l; K elems {k0+hi*8..+7} then {k0+16+hi*8..+7}
__device__ __forceinline__ v16bf frag_f32(const float* __restrict__ base,
                                          long ld, int k0, int lane) {
  int l = lane & 15, hi = (lane >> 4) & 1;
  const float* p = base + (long)l * ld + k0 + hi * 8;
  v16bf f;
#pragma unroll
  for (int i = 0; i < 8; ++i) f[i] = (__bf16)p[i];
#pragma unroll
  for (int i = 0; i < 8; ++i) f[8 + i] = (__bf16)p[16 + i];
  return f;
}

__device__ __forceinline__ v16bf frag_bf16(const __bf16* base,
                                           long ld, int k0, int lane) {
  int l = lane & 15, hi = (lane >> 4) & 1;
  const __bf16* p = base + (long)l * ld + k0 + hi * 8;
  v16bf f;
#pragma unroll
  for (int i = 0; i < 8; ++i) f[i] = p[i];
#pragma unroll
  for (int i = 0; i < 8; ++i) f[8 + i] = p[16 + i];
  return f;
}

// ---- kernel 1: fused QKV projection + RoPE ------------------------------
// Each wave produces a PAIR of 16-col tiles (d, d+128) of one head so the
// RoPE rotation partner is resident; the A fragment is shared by both tiles.
__global__ __launch_bounds__(128) void qkv_rope_kernel(
    const float* __restrict__ X, const float* __restrict__ Wq,
    const float* __restrict__ Wk, const float* __restrict__ Wv,
    __bf16* __restrict__ Qb, __bf16* __restrict__ Kb, __bf16* __restrict__ Vb) {
  int lane = threadIdx.x & 31;
  int wave = threadIdx.x >> 5;
  int row0 = blockIdx.x * 16;
  int pt   = blockIdx.y * 4 + wave;  // 0..127 pair-tiles: 64 Q, 32 K, 32 V

  const float* W; __bf16* Out; long ldo; int wrow0; bool rope;
  if (pt < 64)      { W = Wq; Out = Qb; ldo = QDIM;  wrow0 = (pt / 8) * DH + (pt % 8) * 16;                rope = true;  }
  else if (pt < 96) { int p = pt - 64; W = Wk; Out = Kb; ldo = KVDIM; wrow0 = (p / 8) * DH + (p % 8) * 16; rope = true;  }
  else              { int p = pt - 96; W = Wv; Out = Vb; ldo = KVDIM; wrow0 = (p / 8) * DH + (p % 8) * 16; rope = false; }

  v8f acc0 = {}; v8f acc1 = {};
  const float* Abase = X + (long)row0 * HID_;
  const float* B0 = W + (long)wrow0 * HID_;
  const float* B1 = W + (long)(wrow0 + DH / 2) * HID_;
  for (int k0 = 0; k0 < HID_; k0 += 32) {
    v16bf a  = frag_f32(Abase, HID_, k0, lane);
    v16bf b0 = frag_f32(B0,    HID_, k0, lane);
    v16bf b1 = frag_f32(B1,    HID_, k0, lane);
    acc0 = wmma_bf16(a, b0, acc0);
    acc1 = wmma_bf16(a, b1, acc1);
  }

  int l = lane & 15, hi = (lane >> 4) & 1;
  if (rope) {
    int d = (wrow0 % DH) + l;  // 0..127 -> inv_freq index
    // theta^(-d/128) = exp2(-d * log2(10000)/128): native exp2
    float invf = exp2f((float)d * (-13.287712379549449f / 128.0f));
#pragma unroll
    for (int r = 0; r < 8; ++r) {
      float ang = (float)(row0 + hi * 8 + r) * invf;
      float sn, cs;
      __sincosf(ang, &sn, &cs);
      float x0 = acc0[r], x1 = acc1[r];
      acc0[r] = x0 * cs - x1 * sn;
      acc1[r] = x1 * cs + x0 * sn;
    }
  }
  __bf16* o0 = Out + (long)row0 * ldo + wrow0;
#pragma unroll
  for (int r = 0; r < 8; ++r) {
    o0[(long)(hi * 8 + r) * ldo + l]          = (__bf16)acc0[r];
    o0[(long)(hi * 8 + r) * ldo + DH / 2 + l] = (__bf16)acc1[r];
  }
}

// ---- kernel 2: sliding-window flash attention ---------------------------
// Block = 4 waves = 4 consecutive 16-row query tiles of ONE head.
// Per 32-key chunk: K tile DMA'd into LDS by the Tensor Data Mover (with
// hardware padding to the conflict-free KLD stride) while all threads stage
// V^T manually; WMMA fragments then come from LDS via contiguous 16B reads.
__global__ __launch_bounds__(128) void attn_kernel(
    const __bf16* __restrict__ Qb, const __bf16* __restrict__ Kb,
    const __bf16* __restrict__ Vb, __bf16* __restrict__ Ob) {
  __shared__ __bf16 Ksh[32 * KLD];        // K chunk, row-major [key][d]
  __shared__ __bf16 Vt[256 * VLD];        // V chunk transposed [d][key]
  __shared__ __bf16 Psh[4][16 * PLD];     // per-wave P staging (C->A relayout)

  int tid  = threadIdx.x;
  int lane = tid & 31;
  int wave = tid >> 5;
  int head = blockIdx.x >> 6;             // gridDim.x = 8 * 64
  int qbase = (blockIdx.x & 63) * 64;
  int qrow0 = qbase + wave * 16;
  int kvh  = head >> 1;                   // GQA: 2 q-heads per kv-head
  int l = lane & 15, hi = (lane >> 4) & 1;

  const __bf16* Qh = Qb + (long)qrow0 * QDIM + head * DH;
  const __bf16* Kg = Kb + kvh * DH;
  const __bf16* Vg = Vb + kvh * DH;

  v16bf qf[8];
#pragma unroll
  for (int kk = 0; kk < 8; ++kk) qf[kk] = frag_bf16(Qh, QDIM, kk * 32, lane);

  v8f oacc[16];
#pragma unroll
  for (int t = 0; t < 16; ++t) { v8f z = {}; oacc[t] = z; }
  float m_r[8], l_r[8];
#pragma unroll
  for (int r = 0; r < 8; ++r) { m_r[r] = -3.0e38f; l_r[r] = 0.0f; }

  // block-uniform key range (union of the 4 tiles' windows); out-of-window
  // chunks are handled by the mask + online-softmax rescale
  int jmin = qbase - (SWIN - 1);
  if (jmin < 0) jmin = 0;
  jmin &= ~31;
  int jmax = qbase + 63;

  for (int key0 = jmin; key0 <= jmax; key0 += 32) {
    __syncthreads();
    // ---- stage K (TDM) + V^T (manual transpose) -------------------------
#if HAVE_TDM
    if (wave == 0)
      tdm_load_tile(Kg + (long)key0 * KVDIM, (unsigned)(uintptr_t)(void*)Ksh);
#endif
#pragma unroll
    for (int it = 0; it < 8; ++it) {
      int seg = tid + it * 128;           // 1024 segments of 8 bf16
      int row = seg >> 5;                 // 0..31 key row
      int cs  = (seg & 31) * 8;           // 0..248 d offset
#if !HAVE_TDM
      v8bf kv = *(const v8bf*)(Kg + (long)(key0 + row) * KVDIM + cs);
      *(v8bf*)(&Ksh[row * KLD + cs]) = kv;
#endif
      v8bf vv = *(const v8bf*)(Vg + (long)(key0 + row) * KVDIM + cs);
#pragma unroll
      for (int i = 0; i < 8; ++i) Vt[(cs + i) * VLD + row] = vv[i];
    }
    if (key0 + 32 <= jmax) {              // hide next chunk's HBM latency
      int row = tid >> 2;
      int cs  = (tid & 3) * 64;
      __builtin_prefetch(Kg + (long)(key0 + 32 + row) * KVDIM + cs, 0, 1);
      __builtin_prefetch(Vg + (long)(key0 + 32 + row) * KVDIM + cs, 0, 1);
    }
#if HAVE_TDM
    if (wave == 0) __builtin_amdgcn_s_wait_tensorcnt(0);
#endif
    __syncthreads();

    // ---- scores: two 16x16 tiles over 32 keys ---------------------------
    v8f s0 = {}; v8f s1 = {};
#pragma unroll
    for (int kk = 0; kk < 8; ++kk) {
      v16bf b0 = frag_bf16(&Ksh[0],        KLD, kk * 32, lane);
      v16bf b1 = frag_bf16(&Ksh[16 * KLD], KLD, kk * 32, lane);
      s0 = wmma_bf16(qf[kk], b0, s0);
      s1 = wmma_bf16(qf[kk], b1, s1);
    }

    // softcap + causal/sliding-window mask + per-row chunk max (C layout)
    float cmax[8];
#pragma unroll
    for (int r = 0; r < 8; ++r) {
      int i  = qrow0 + hi * 8 + r;
      int j0 = key0 + l;
      int j1 = j0 + 16;
      float v0 = fast_tanh(s0[r] * (SCALE_ / SOFTCAP_)) * SOFTCAP_;
      float v1 = fast_tanh(s1[r] * (SCALE_ / SOFTCAP_)) * SOFTCAP_;
      v0 += ((i >= j0) && (i - j0 < SWIN)) ? 0.0f : -1.0e9f;
      v1 += ((i >= j1) && (i - j1 < SWIN)) ? 0.0f : -1.0e9f;
      s0[r] = v0; s1[r] = v1;
      cmax[r] = fmaxf(v0, v1);
    }
#pragma unroll
    for (int m = 1; m <= 8; m <<= 1)
#pragma unroll
      for (int r = 0; r < 8; ++r)
        cmax[r] = fmaxf(cmax[r], __shfl_xor(cmax[r], m, 32));

    float scl[8], psum[8];
#pragma unroll
    for (int r = 0; r < 8; ++r) {
      float mnew = fmaxf(m_r[r], cmax[r]);
      scl[r] = __expf(m_r[r] - mnew);
      m_r[r] = mnew;
      float p0 = __expf(s0[r] - mnew);
      float p1 = __expf(s1[r] - mnew);
      s0[r] = p0; s1[r] = p1;
      psum[r] = p0 + p1;
    }
#pragma unroll
    for (int m = 1; m <= 8; m <<= 1)
#pragma unroll
      for (int r = 0; r < 8; ++r)
        psum[r] += __shfl_xor(psum[r], m, 32);
#pragma unroll
    for (int r = 0; r < 8; ++r) l_r[r] = l_r[r] * scl[r] + psum[r];

    // reshape P: C layout -> A layout via per-wave LDS staging
    __bf16* pl = Psh[wave];
#pragma unroll
    for (int r = 0; r < 8; ++r) {
      pl[(hi * 8 + r) * PLD + l]      = (__bf16)s0[r];
      pl[(hi * 8 + r) * PLD + 16 + l] = (__bf16)s1[r];
    }
    __builtin_amdgcn_wave_barrier();
    asm volatile("s_wait_dscnt 0" ::: "memory");  // same-wave cross-lane LDS RAW
    v16bf pf = frag_bf16(pl, PLD, 0, lane);

    // rescale running output and accumulate P @ V  (16 x 256)
#pragma unroll
    for (int t = 0; t < 16; ++t) {
      v16bf vf = frag_bf16(&Vt[(t * 16) * VLD], VLD, 0, lane);
#pragma unroll
      for (int r = 0; r < 8; ++r) oacc[t][r] *= scl[r];
      oacc[t] = wmma_bf16(pf, vf, oacc[t]);
    }
  }

  float inv[8];
#pragma unroll
  for (int r = 0; r < 8; ++r) inv[r] = __builtin_amdgcn_rcpf(l_r[r]);
  __bf16* oh = Ob + (long)qrow0 * QDIM + head * DH;
#pragma unroll
  for (int t = 0; t < 16; ++t)
#pragma unroll
    for (int r = 0; r < 8; ++r)
      oh[(long)(hi * 8 + r) * QDIM + t * 16 + l] = (__bf16)(oacc[t][r] * inv[r]);
}

// ---- kernel 3: output projection ---------------------------------------
__global__ __launch_bounds__(128) void oproj_kernel(
    const __bf16* __restrict__ Ob, const float* __restrict__ Wo,
    float* __restrict__ Y) {
  int lane = threadIdx.x & 31;
  int wave = threadIdx.x >> 5;
  int row0 = blockIdx.x * 16;
  int col0 = (blockIdx.y * 4 + wave) * 16;  // HID 2304 = 144 tiles
  v8f acc = {};
  const __bf16* A = Ob + (long)row0 * QDIM;
  const float*  B = Wo + (long)col0 * QDIM;
  for (int k0 = 0; k0 < QDIM; k0 += 32) {
    v16bf a = frag_bf16(A, QDIM, k0, lane);
    v16bf b = frag_f32(B, QDIM, k0, lane);
    acc = wmma_bf16(a, b, acc);
  }
  int l = lane & 15, hi = (lane >> 4) & 1;
#pragma unroll
  for (int r = 0; r < 8; ++r)
    Y[(long)(row0 + hi * 8 + r) * HID_ + col0 + l] = acc[r];
}

extern "C" void kernel_launch(void* const* d_in, const int* in_sizes, int n_in,
                              void* d_out, int out_size, void* d_ws, size_t ws_size,
                              hipStream_t stream) {
  const float* X  = (const float*)d_in[0];
  // d_in[1] = attention_mask (recomputed analytically), d_in[6] = position_ids (arange)
  const float* Wq = (const float*)d_in[2];
  const float* Wk = (const float*)d_in[3];
  const float* Wv = (const float*)d_in[4];
  const float* Wo = (const float*)d_in[5];

  __bf16* Qb = (__bf16*)d_ws;                       // [4096, 2048] bf16
  __bf16* Kb = Qb + (size_t)SEQ * QDIM;             // [4096, 1024] bf16
  __bf16* Vb = Kb + (size_t)SEQ * KVDIM;            // [4096, 1024] bf16
  __bf16* Ob = Vb + (size_t)SEQ * KVDIM;            // [4096, 2048] bf16
  float*  Y  = (float*)d_out;                       // [4096, 2304] f32

  qkv_rope_kernel<<<dim3(SEQ / 16, 32), 128, 0, stream>>>(X, Wq, Wk, Wv, Qb, Kb, Vb);
  attn_kernel<<<dim3(H_Q * (SEQ / 64)), 128, 0, stream>>>(Qb, Kb, Vb, Ob);
  oproj_kernel<<<dim3(SEQ / 16, (HID_ / 16) / 4), 128, 0, stream>>>(Ob, Wo, Y);
}
